// MLPPredictor_40982577938826
// MI455X (gfx1250) — matrix operand order repaired and checked by reading, not compiled, not run on hardware
//
#include <hip/hip_runtime.h>
#include <hip/hip_bf16.h>

typedef __attribute__((ext_vector_type(16))) __bf16        v16bf;
typedef __attribute__((ext_vector_type(8)))  float         v8f;
typedef __attribute__((ext_vector_type(8)))  unsigned int  v8u;

#define IN_F 256   // input features per node
#define HID  64    // hidden size
#define COLS 128   // concatenated [A|B] columns of precomputed C
#define WF_ELEMS (8*8*32*16)   // [ntile][kstep][lane][16 bf16]
#define LDS_ROW_DW 132         // 128 dwords (256 bf16) + 4 dword pad -> conflict-free

__device__ __forceinline__ unsigned pk_bf16(float lo, float hi) {
  // two f32 -> packed bf16 pair (truncate), lo in [15:0], hi in [31:16]
  return (__builtin_bit_cast(unsigned, lo) >> 16) |
         (__builtin_bit_cast(unsigned, hi) & 0xFFFF0000u);
}

// Swizzle W1[512,64] into WMMA B-fragment layout for Wcat[256,128]:
//   Wcat[k][n] = (n<64) ? W1[k][n] : W1[256+k][n-64]
// B frag order used here (contiguous k, as A): lane<16 -> K 0..15, lane>=16 -> K 16..31; col = lane&15.
// WF index = ((nt*8 + ks)*32 + lane)*16 + j
__global__ void prep_w_kernel(const float* __restrict__ W1,
                              unsigned short* __restrict__ WF) {
  int idx = blockIdx.x * blockDim.x + threadIdx.x;
  if (idx >= WF_ELEMS) return;
  int j    = idx & 15;
  int lane = (idx >> 4) & 31;
  int ks   = (idx >> 9) & 7;
  int nt   = idx >> 12;
  int k = ks * 32 + ((lane < 16) ? 0 : 16) + j;
  int n = nt * 16 + (lane & 15);
  float val = (n < HID) ? W1[k * HID + n] : W1[(IN_F + k) * HID + (n - HID)];
  WF[idx] = (unsigned short)(__builtin_bit_cast(unsigned, val) >> 16);
}

// C[n_nodes,128] = bf16(h) @ bf16(Wcat), f32 accumulate.
// 8 waves/block; wave w owns N-tile w with B stationary in 64 VGPRs.
// A tile staged f32->bf16 into LDS (coalesced); all 16 ds_loads issued ahead of
// the WMMA chain so waits are progressive; grid-stride over M tiles.
__global__ void __launch_bounds__(256)
node_gemm_kernel(const float* __restrict__ h,
                 const unsigned short* __restrict__ WF,
                 float* __restrict__ C, int n_nodes, int mtiles) {
  __shared__ unsigned As[16 * LDS_ROW_DW];   // 16 rows of bf16(256) + pad

  const int tid  = threadIdx.x;
  const int lane = tid & 31;
  const int wave = tid >> 5;                 // == N tile index (0..7)

  // ---- stationary B fragments for this wave's N tile (loaded once) ----
  v16bf Bfrag[8];
#pragma unroll
  for (int ks = 0; ks < 8; ++ks) {
    const uint4* bp =
        (const uint4*)(WF + ((((wave * 8) + ks) * 32 + lane) << 4));
    uint4 b0 = bp[0], b1 = bp[1];
    v8u bu = {b0.x, b0.y, b0.z, b0.w, b1.x, b1.y, b1.z, b1.w};
    Bfrag[ks] = __builtin_bit_cast(v16bf, bu);
  }

  const int ldsrow = tid >> 4;           // 0..15 : row this thread stages
  const int ldscol = (tid & 15) * 16;    // 16 consecutive f32 per thread
  const int klo    = (lane < 16) ? 0 : 16;
  const int arow   = lane & 15;

  v8f zero = {0.f, 0.f, 0.f, 0.f, 0.f, 0.f, 0.f, 0.f};

  for (int mt = blockIdx.x; mt < mtiles; mt += gridDim.x) {
    const long m0 = (long)mt * 16;

    // ---- cooperative coalesced stage: h tile f32 -> LDS bf16 ----
    long srow = m0 + ldsrow;
    if (srow >= n_nodes) srow = n_nodes - 1;   // clamp; stores guarded below
    const float* hp = h + srow * IN_F + ldscol;
    float4 f0 = *(const float4*)(hp);
    float4 f1 = *(const float4*)(hp + 4);
    float4 f2 = *(const float4*)(hp + 8);
    float4 f3 = *(const float4*)(hp + 12);
    uint4 lo = {pk_bf16(f0.x, f0.y), pk_bf16(f0.z, f0.w),
                pk_bf16(f1.x, f1.y), pk_bf16(f1.z, f1.w)};
    uint4 hi = {pk_bf16(f2.x, f2.y), pk_bf16(f2.z, f2.w),
                pk_bf16(f3.x, f3.y), pk_bf16(f3.z, f3.w)};
    unsigned* dst = As + ldsrow * LDS_ROW_DW + (tid & 15) * 8;
    *(uint4*)(dst)     = lo;
    *(uint4*)(dst + 4) = hi;
    __syncthreads();

    // ---- issue ALL A-fragment ds_loads, then chain the 8 WMMAs ----
    // A frag order used here (contiguous k, same as B): lane<16 holds K {0..15},
    // lane>=16 holds K {16..31}; row = lane&15.
    const unsigned short* Ab = (const unsigned short*)As;
    const unsigned short* abase = Ab + arow * (2 * LDS_ROW_DW) + klo;
    uint4 a0[8], a1[8];
#pragma unroll
    for (int ks = 0; ks < 8; ++ks) {
      a0[ks] = *(const uint4*)(abase + ks * 32);
      a1[ks] = *(const uint4*)(abase + ks * 32 + 8);
    }

    v8f acc = zero;
#pragma unroll
    for (int ks = 0; ks < 8; ++ks) {
      v8u au = {a0[ks].x, a0[ks].y, a0[ks].z, a0[ks].w,
                a1[ks].x, a1[ks].y, a1[ks].z, a1[ks].w};
      v16bf a = __builtin_bit_cast(v16bf, au);
      acc = __builtin_amdgcn_wmma_f32_16x16x32_bf16(
          false, a, false, Bfrag[ks], (short)0, acc, false, false);
    }

    // C/D layout: VGPR r, lanes 0-15 -> M=r, lanes 16-31 -> M=r+8; N=lane&15
    if (m0 + 16 <= (long)n_nodes) {
      // fast path: unguarded stores, one base address + immediate offsets
      float* cp = C + (m0 + ((lane < 16) ? 0 : 8)) * COLS +
                  wave * 16 + (lane & 15);
#pragma unroll
      for (int r = 0; r < 8; ++r) cp[r * COLS] = acc[r];
    } else {
#pragma unroll
      for (int r = 0; r < 8; ++r) {
        long orow = m0 + ((lane < 16) ? r : r + 8);
        if (orow < n_nodes)
          C[orow * COLS + wave * 16 + (lane & 15)] = acc[r];
      }
    }
    __syncthreads();   // LDS reuse across grid-stride iterations
  }
}

// out[e] = relu(C[u][0:64] + C[v][64:128] + b1) . W2 + b2
__global__ void edge_mlp_kernel(const int* __restrict__ ei,
                                const float* __restrict__ C,
                                const float* __restrict__ b1,
                                const float* __restrict__ W2,
                                const float* __restrict__ b2,
                                float* __restrict__ out, int E) {
  int e = blockIdx.x * blockDim.x + threadIdx.x;
  if (e >= E) return;
  long u = ei[e];
  long v = ei[E + e];
  const float* Ar = C + u * COLS;        // h_u @ W1_top part
  const float* Br = C + v * COLS + HID;  // h_v @ W1_bot part
  float sum = b2[0];
#pragma unroll
  for (int i = 0; i < HID; i += 4) {
    float4 a    = *(const float4*)(Ar + i);
    float4 bb   = *(const float4*)(Br + i);
    float4 bias = *(const float4*)(b1 + i);
    float4 w    = *(const float4*)(W2 + i);
    sum += fmaxf(a.x + bb.x + bias.x, 0.f) * w.x;
    sum += fmaxf(a.y + bb.y + bias.y, 0.f) * w.y;
    sum += fmaxf(a.z + bb.z + bias.z, 0.f) * w.z;
    sum += fmaxf(a.w + bb.w + bias.w, 0.f) * w.w;
  }
  out[e] = sum;
}

extern "C" void kernel_launch(void* const* d_in, const int* in_sizes, int n_in,
                              void* d_out, int out_size, void* d_ws,
                              size_t ws_size, hipStream_t stream) {
  const float* h  = (const float*)d_in[0];
  const int*   ei = (const int*)d_in[1];
  const float* W1 = (const float*)d_in[2];
  const float* b1 = (const float*)d_in[3];
  const float* W2 = (const float*)d_in[4];
  const float* b2 = (const float*)d_in[5];
  float* out = (float*)d_out;

  int n_nodes = in_sizes[0] / IN_F;
  int E       = in_sizes[1] / 2;

  unsigned short* WF = (unsigned short*)d_ws;                       // 64 KB
  float* C = (float*)((char*)d_ws + WF_ELEMS * sizeof(unsigned short));
  // C needs n_nodes * 128 * 4 bytes (~51.2 MB for 100k nodes)

  prep_w_kernel<<<(WF_ELEMS + 255) / 256, 256, 0, stream>>>(W1, WF);

  int mtiles = (n_nodes + 15) / 16;
  int blocks = (mtiles + 3) / 4;     // ~4 M-tiles per block amortize B load
  node_gemm_kernel<<<blocks, 256, 0, stream>>>(h, WF, C, n_nodes, mtiles);

  edge_mlp_kernel<<<(E + 255) / 256, 256, 0, stream>>>(ei, C, b1, W2, b2, out, E);
}
